// GraphSAGEEncoder_57389353009169
// MI455X (gfx1250) — matrix operand order repaired and checked by reading, not compiled, not run on hardware
//
#include <hip/hip_runtime.h>
#include <hip/hip_bf16.h>

#define NNODES 50000
#define NEDGES 800000
#define DIN    128
#define DHID   128
#define DOUT   64
#define BN_EPS 1e-5f

typedef __attribute__((ext_vector_type(16))) __bf16        v16bf;
typedef __attribute__((ext_vector_type(8)))  float         v8f;
typedef __attribute__((ext_vector_type(4)))  unsigned int  v4u;
typedef __attribute__((ext_vector_type(8)))  int           v8i;
typedef __attribute__((ext_vector_type(4)))  int           v4i;

// ---------- helpers ----------
__device__ __forceinline__ unsigned short f2bf(float f) {
    unsigned int u = __float_as_uint(f);
    u += 0x7fffu + ((u >> 16) & 1u);   // round-to-nearest-even
    return (unsigned short)(u >> 16);
}

// Build a D# for a 2D bf16 tile [rows x 128] (row stride 128) and issue a
// Tensor Data Mover load into LDS.  Wave-uniform arguments -> SGPR groups.
__device__ __forceinline__ void tdm_load_weights(const unsigned short* gsrc,
                                                 unsigned short* ldst, int rows) {
    unsigned long long ga = (unsigned long long)gsrc;
    unsigned lds = (unsigned)(unsigned long long)ldst;      // low 32b = LDS byte addr
    v4u g0;
    g0.x = 1u;                                              // count=1, user descriptor
    g0.y = lds;                                             // lds_addr
    g0.z = (unsigned)ga;                                    // global_addr[31:0]
    g0.w = ((unsigned)(ga >> 32) & 0x01FFFFFFu) | (2u << 30); // ga[56:32] | type=2
    v8i g1;
    g1[0] = 0x00010000;                                     // data_size=2B, mask/flags=0
    g1[1] = (int)(((unsigned)DIN & 0xFFFFu) << 16);         // tensor_dim0=128 @ bits 63:48
    g1[2] = (int)(((unsigned)rows & 0xFFFFu) << 16);        // tensor_dim1=rows @ bits 95:80
    g1[3] = (int)(((unsigned)DIN) << 16);                   // tile_dim0=128 @ bits 127:112
    g1[4] = rows;                                           // tile_dim1=rows, tile_dim2=0
    g1[5] = DIN;                                            // tensor_dim0_stride lo32
    g1[6] = 0;                                              // stride0 hi | stride1 lo
    g1[7] = 0;                                              // stride1 hi
    v4i gz4 = (v4i)0;                                       // groups 2/3 unused (2D tile)
    v8i gz8 = (v8i)0;
    __builtin_amdgcn_tensor_load_to_lds(g0, g1, gz4, gz4, gz8, 0);
}

// ---------- generic zero ----------
__global__ void zero_f32(float* __restrict__ p, int n) {
    int i = blockIdx.x * blockDim.x + threadIdx.x;
    if (i < n) p[i] = 0.0f;
}

// ---------- fp32 -> bf16 cast (n multiple of 4) ----------
__global__ void cast_bf16(const float* __restrict__ X, unsigned short* __restrict__ Y, int n4) {
    int i = blockIdx.x * blockDim.x + threadIdx.x;
    if (i >= n4) return;
    const float4 v = *(const float4*)(X + 4 * i);
    ushort4 o;
    o.x = f2bf(v.x); o.y = f2bf(v.y); o.z = f2bf(v.z); o.w = f2bf(v.w);
    *(ushort4*)(Y + 4 * i) = o;
}

// ---------- weight cast + transpose:  Wt[n*K + k] = bf16(W[k*Nout + n]) ----------
__global__ void wcast_t(const float* __restrict__ W, unsigned short* __restrict__ Wt,
                        int K, int Nout) {
    int id = blockIdx.x * blockDim.x + threadIdx.x;
    if (id >= K * Nout) return;
    int n = id / K, k = id % K;
    Wt[id] = f2bf(W[k * Nout + n]);
}

// ---------- edge scatter-add: Agg[dst] += X[src], 4 channels / thread ----------
__global__ void scatter_add(const float* __restrict__ X, const int* __restrict__ ei,
                            float* __restrict__ Agg, int E) {
    int idx = blockIdx.x * blockDim.x + threadIdx.x;       // E * 32 threads (D/4 = 32)
    if (idx >= E * (DIN / 4)) return;
    int e = idx >> 5;
    int c = (idx & 31) << 2;
    int s = ei[e];
    int d = ei[E + e];
    const float* xp = X + (size_t)s * DIN + c;
    __builtin_prefetch(Agg + (size_t)d * DIN + c, 1, 1);   // global_prefetch_b8
    const float4 v = *(const float4*)xp;
    float* p = Agg + (size_t)d * DIN + c;
    atomicAdd(p + 0, v.x);
    atomicAdd(p + 1, v.y);
    atomicAdd(p + 2, v.z);
    atomicAdd(p + 3, v.w);
}

// ---------- fused SAGE GEMM:  H = Ab@Wl + Xb@Wr + bias  (bf16 WMMA, f32 acc) ----------
// One 16-row strip per wave; all NT column tiles accumulated in registers.
// Weights staged into LDS by the Tensor Data Mover (wave 0), read as ds_load_b128.
template<int NT>
__global__ __launch_bounds__(256)
void sage_gemm(const unsigned short* __restrict__ Ab, const unsigned short* __restrict__ Xb,
               const unsigned short* __restrict__ Wlt, const unsigned short* __restrict__ Wrt,
               const float* __restrict__ bias, float* __restrict__ H, int Mtiles) {
    constexpr int Dout = NT * 16;
    __shared__ unsigned short sW[2 * Dout * DIN];          // [Wl | Wr], 32/64 KB
    unsigned short* sWl = sW;
    unsigned short* sWr = sW + Dout * DIN;

    const int wave = threadIdx.x >> 5;
    const int lane = threadIdx.x & 31;

    if (threadIdx.x < 32) {                                // wave 0 drives the TDM
        tdm_load_weights(Wlt, sWl, Dout);
        tdm_load_weights(Wrt, sWr, Dout);
        __builtin_amdgcn_s_wait_tensorcnt(0);
    }
    __syncthreads();

    const int tm = blockIdx.x * 8 + wave;
    if (tm >= Mtiles) return;                              // wave-uniform exit

    const int half = lane >> 4;
    const int lr   = lane & 15;

    v8f acc[NT];
#pragma unroll
    for (int nt = 0; nt < NT; ++nt) {
        const float bv = bias[nt * 16 + lr];
#pragma unroll
        for (int r = 0; r < 8; ++r) acc[nt][r] = bv;
    }

    const unsigned short* arow = Ab + (size_t)(tm * 16 + lr) * DIN;
    const unsigned short* xrow = Xb + (size_t)(tm * 16 + lr) * DIN;

    union Frag { uint4 u[2]; v16bf v; };

#pragma unroll
    for (int k0 = 0; k0 < DIN; k0 += 32) {
        Frag a, x;
        // A fragment: elems 0..7 -> K = k0+8*half+i ; elems 8..15 -> K = k0+16+8*half+i
        a.u[0] = *(const uint4*)(arow + k0 + 8 * half);
        a.u[1] = *(const uint4*)(arow + k0 + 16 + 8 * half);
        x.u[0] = *(const uint4*)(xrow + k0 + 8 * half);
        x.u[1] = *(const uint4*)(xrow + k0 + 16 + 8 * half);
#pragma unroll
        for (int nt = 0; nt < NT; ++nt) {
            // B fragment: elems 0..7 -> K = k0+8*half+i ; elems 8..15 -> K = k0+16+8*half+i (rows of transposed W in LDS)
            const unsigned short* wl = sWl + (nt * 16 + lr) * DIN + k0 + 8 * half;
            const unsigned short* wr = sWr + (nt * 16 + lr) * DIN + k0 + 8 * half;
            Frag bl, br;
            bl.u[0] = *(const uint4*)(wl);
            bl.u[1] = *(const uint4*)(wl + 16);
            br.u[0] = *(const uint4*)(wr);
            br.u[1] = *(const uint4*)(wr + 16);
            acc[nt] = __builtin_amdgcn_wmma_f32_16x16x32_bf16(false, a.v, false, bl.v,
                                                              (short)0, acc[nt], false, false);
            acc[nt] = __builtin_amdgcn_wmma_f32_16x16x32_bf16(false, x.v, false, br.v,
                                                              (short)0, acc[nt], false, false);
        }
    }

    float* outb = H + (size_t)(tm * 16) * Dout + lr;
#pragma unroll
    for (int nt = 0; nt < NT; ++nt)
#pragma unroll
        for (int r = 0; r < 8; ++r)
            outb[(size_t)(r + 8 * half) * Dout + nt * 16] = acc[nt][r];  // M=r+8*half, N=lane%16
}

// ---------- BN stats: per-channel sum / sumsq ----------
#define STAT_ROWS 250
__global__ void bn_stats(const float* __restrict__ H, float* __restrict__ stats, int Nrows) {
    const int c  = threadIdx.x;                     // blockDim.x == 128
    const int r0 = blockIdx.x * STAT_ROWS;
    const int r1 = min(r0 + STAT_ROWS, Nrows);
    float s = 0.f, sq = 0.f;
    for (int r = r0; r < r1; ++r) {
        float v = H[(size_t)r * DHID + c];
        s += v; sq += v * v;
    }
    atomicAdd(&stats[c], s);
    atomicAdd(&stats[DHID + c], sq);
}

// ---------- BN normalize + exact GELU (in place) ----------
__global__ void bn_gelu(float* __restrict__ H, const float* __restrict__ stats,
                        const float* __restrict__ g, const float* __restrict__ be,
                        int n, float invN) {
    int i = blockIdx.x * blockDim.x + threadIdx.x;
    if (i >= n) return;
    int c = i & (DHID - 1);
    float mu  = stats[c] * invN;
    float var = stats[DHID + c] * invN - mu * mu;
    float xh  = (H[i] - mu) * rsqrtf(var + BN_EPS) * g[c] + be[c];
    H[i] = 0.5f * xh * (1.0f + erff(xh * 0.70710678118654752f));
}

// =====================================================================
extern "C" void kernel_launch(void* const* d_in, const int* in_sizes, int n_in,
                              void* d_out, int out_size, void* d_ws, size_t ws_size,
                              hipStream_t stream) {
    const float* x   = (const float*)d_in[0];
    const int*   ei  = (const int*)  d_in[1];
    const float* W0l = (const float*)d_in[2];
    const float* W0r = (const float*)d_in[3];
    const float* b0  = (const float*)d_in[4];
    const float* g0  = (const float*)d_in[5];
    const float* be0 = (const float*)d_in[6];
    const float* W1l = (const float*)d_in[7];
    const float* W1r = (const float*)d_in[8];
    const float* b1  = (const float*)d_in[9];
    const float* g1  = (const float*)d_in[10];
    const float* be1 = (const float*)d_in[11];
    const float* W2l = (const float*)d_in[12];
    const float* W2r = (const float*)d_in[13];
    const float* b2  = (const float*)d_in[14];
    float* out = (float*)d_out;

    // ---- workspace carve-out ----
    char* ws = (char*)d_ws;
    size_t o = 0;
    auto take = [&](size_t bytes) { char* p = ws + o; o += (bytes + 255) & ~(size_t)255; return p; };
    float*          agg   = (float*)         take((size_t)NNODES * DHID * 4);
    float*          h0    = (float*)         take((size_t)NNODES * DHID * 4);
    float*          h1    = (float*)         take((size_t)NNODES * DHID * 4);
    unsigned short* xb    = (unsigned short*)take((size_t)NNODES * DHID * 2);
    unsigned short* ab    = (unsigned short*)take((size_t)NNODES * DHID * 2);
    unsigned short* w0lt  = (unsigned short*)take(DIN  * DHID * 2);
    unsigned short* w0rt  = (unsigned short*)take(DIN  * DHID * 2);
    unsigned short* w1lt  = (unsigned short*)take(DHID * DHID * 2);
    unsigned short* w1rt  = (unsigned short*)take(DHID * DHID * 2);
    unsigned short* w2lt  = (unsigned short*)take(DHID * DOUT * 2);
    unsigned short* w2rt  = (unsigned short*)take(DHID * DOUT * 2);
    float*          stats = (float*)         take(2 * DHID * 4);

    const int ND   = NNODES * DHID;                 // 6.4M
    const int ND4  = ND / 4;
    const int scatterThreads = NEDGES * (DIN / 4);  // 25.6M
    const int Mtiles = NNODES / 16;                 // 3125
    const float invN = 1.0f / (float)NNODES;

    dim3 B256(256);

    // ---- weights: cast + transpose (every call; deterministic) ----
    wcast_t<<<(DIN * DHID + 255) / 256, B256, 0, stream>>>(W0l, w0lt, DIN, DHID);
    wcast_t<<<(DIN * DHID + 255) / 256, B256, 0, stream>>>(W0r, w0rt, DIN, DHID);
    wcast_t<<<(DHID * DHID + 255) / 256, B256, 0, stream>>>(W1l, w1lt, DHID, DHID);
    wcast_t<<<(DHID * DHID + 255) / 256, B256, 0, stream>>>(W1r, w1rt, DHID, DHID);
    wcast_t<<<(DHID * DOUT + 255) / 256, B256, 0, stream>>>(W2l, w2lt, DHID, DOUT);
    wcast_t<<<(DHID * DOUT + 255) / 256, B256, 0, stream>>>(W2r, w2rt, DHID, DOUT);

    // ================= layer 0 =================
    cast_bf16<<<(ND4 + 255) / 256, B256, 0, stream>>>(x, xb, ND4);
    zero_f32<<<(ND + 255) / 256, B256, 0, stream>>>(agg, ND);
    zero_f32<<<1, B256, 0, stream>>>(stats, 2 * DHID);
    scatter_add<<<(scatterThreads + 255) / 256, B256, 0, stream>>>(x, ei, agg, NEDGES);
    cast_bf16<<<(ND4 + 255) / 256, B256, 0, stream>>>(agg, ab, ND4);
    sage_gemm<8><<<(Mtiles + 7) / 8, B256, 0, stream>>>(ab, xb, w0lt, w0rt, b0, h0, Mtiles);
    bn_stats<<<(NNODES + STAT_ROWS - 1) / STAT_ROWS, dim3(DHID), 0, stream>>>(h0, stats, NNODES);
    bn_gelu<<<(ND + 255) / 256, B256, 0, stream>>>(h0, stats, g0, be0, ND, invN);

    // ================= layer 1 =================
    cast_bf16<<<(ND4 + 255) / 256, B256, 0, stream>>>(h0, xb, ND4);
    zero_f32<<<(ND + 255) / 256, B256, 0, stream>>>(agg, ND);
    zero_f32<<<1, B256, 0, stream>>>(stats, 2 * DHID);
    scatter_add<<<(scatterThreads + 255) / 256, B256, 0, stream>>>(h0, ei, agg, NEDGES);
    cast_bf16<<<(ND4 + 255) / 256, B256, 0, stream>>>(agg, ab, ND4);
    sage_gemm<8><<<(Mtiles + 7) / 8, B256, 0, stream>>>(ab, xb, w1lt, w1rt, b1, h1, Mtiles);
    bn_stats<<<(NNODES + STAT_ROWS - 1) / STAT_ROWS, dim3(DHID), 0, stream>>>(h1, stats, NNODES);
    bn_gelu<<<(ND + 255) / 256, B256, 0, stream>>>(h1, stats, g1, be1, ND, invN);

    // ================= layer 2 (output, no BN/GELU) =================
    cast_bf16<<<(ND4 + 255) / 256, B256, 0, stream>>>(h1, xb, ND4);
    zero_f32<<<(ND + 255) / 256, B256, 0, stream>>>(agg, ND);
    scatter_add<<<(scatterThreads + 255) / 256, B256, 0, stream>>>(h1, ei, agg, NEDGES);
    cast_bf16<<<(ND4 + 255) / 256, B256, 0, stream>>>(agg, ab, ND4);
    sage_gemm<4><<<(Mtiles + 7) / 8, B256, 0, stream>>>(ab, xb, w2lt, w2rt, b2, out, Mtiles);
}
